// MultiScaleAttention_85770496901417
// MI455X (gfx1250) — matrix operand order repaired and checked by reading, not compiled, not run on hardware
//
#include <hip/hip_runtime.h>
#include <hip/hip_bf16.h>

typedef __attribute__((ext_vector_type(16))) _Float16 v16h;
typedef __attribute__((ext_vector_type(8)))  float    v8f;

#define B_     16
#define N_TOK  1025
#define DIM_   768
#define NH_    12
#define HD_    64
#define KV_N   257
#define M_ROWS (B_*N_TOK)   // 16400

// ---------------------------------------------------------------------------
// WMMA fragment loaders (wave32, 16x16x32 f16; layouts per CDNA5 ISA 7.12.2)
// A (16xK=32): lane l -> row m=l%16; elem e -> K = (e/8)*16 + (l>=16?8:0) + e%8
//   => two contiguous 8-half runs per lane from row-major LDS.
// B (K=32x16): lane l -> col n=l%16; elem e -> K = (e/8)*16 + (l>=16?8:0) + e%8
//   => two contiguous 8-half runs per lane from N-major (transposed) LDS.
// ---------------------------------------------------------------------------
__device__ __forceinline__ v16h load_a_frag(const _Float16* base, int stride, int koff) {
  int l   = threadIdx.x & 31;
  int h16 = (l >> 4) & 1;
  int m   = l & 15;
  const _Float16* p = base + m * stride + koff + h16 * 8;
  v16h a;
#pragma unroll
  for (int e = 0; e < 8; ++e) { a[e] = p[e]; a[e + 8] = p[e + 16]; }
  return a;
}

__device__ __forceinline__ v16h load_b_frag(const _Float16* base, int stride, int koff) {
  int l   = threadIdx.x & 31;
  int h16 = (l >> 4) & 1;
  int n   = l & 15;
  const _Float16* p = base + n * stride + koff + h16 * 8;
  v16h b;
#pragma unroll
  for (int e = 0; e < 8; ++e) { b[e] = p[e]; b[e + 8] = p[e + 16]; }
  return b;
}

// ---------------------------------------------------------------------------
// Generic f32->f16 WMMA GEMM, 128x64 tile, 256 threads (8 waves), K step 32.
// MODE 0: C cols [0,1536) split into K (cols<768) / V buffers, [B,nH,N,hd] layout
// MODE 1: C cols [0,768) into per-head [B,nH,N,hd] layout
// MODE 2: plain row-major + bias (projection, writes d_out)
// ---------------------------------------------------------------------------
template <int MODE>
__global__ __launch_bounds__(256) void wmma_gemm(
    const float* __restrict__ A, const float* __restrict__ W,
    const float* __restrict__ bias, float* __restrict__ out0,
    float* __restrict__ out1, int M, int N, int K, int ldw, int wcol0)
{
  __shared__ _Float16 As[128][40];  // 128x32 tile, padded stride
  __shared__ _Float16 Bs[64][40];   // B stored transposed: [n][k]

  int tid     = threadIdx.x;
  int wvid    = tid >> 5;
  int rowBase = blockIdx.x * 128;
  int nBase   = blockIdx.y * 64;

  v8f acc[4];
#pragma unroll
  for (int j = 0; j < 4; ++j)
#pragma unroll
    for (int e = 0; e < 8; ++e) acc[j][e] = 0.f;

  int nk = K >> 5;
  for (int kt = 0; kt < nk; ++kt) {
    int k0 = kt << 5;
    // stage A 128x32 (convert f32 -> f16)
#pragma unroll
    for (int t = 0; t < 16; ++t) {
      int idx  = tid + t * 256;
      int r    = idx >> 5, c = idx & 31;
      int grow = rowBase + r;
      float v  = (grow < M) ? A[(size_t)grow * K + k0 + c] : 0.f;
      As[r][c] = (_Float16)v;
    }
    // stage B 32x64, transposed into [n][k]
#pragma unroll
    for (int t = 0; t < 8; ++t) {
      int idx = tid + t * 256;
      int kr  = idx >> 6, nc = idx & 63;
      float v = W[(size_t)(k0 + kr) * ldw + wcol0 + nBase + nc];
      Bs[nc][kr] = (_Float16)v;
    }
    if (kt + 1 < nk) {  // hint next A tile into cache
      int grow = rowBase + (tid >> 5);
      if (grow < M) __builtin_prefetch(&A[(size_t)grow * K + k0 + 32], 0, 1);
    }
    __syncthreads();

    v16h a = load_a_frag(&As[wvid * 16][0], 40, 0);
#pragma unroll
    for (int j = 0; j < 4; ++j) {
      v16h b = load_b_frag(&Bs[j * 16][0], 40, 0);
      acc[j] = __builtin_amdgcn_wmma_f32_16x16x32_f16(
          false, a, false, b, (short)0, acc[j], false, false);
    }
    __syncthreads();
  }

  // store (C layout: lane l holds col l%16; VGPR r holds row r + 8*(l>=16))
  int l    = tid & 31;
  int h16  = (l >> 4) & 1;
  int col0 = l & 15;
#pragma unroll
  for (int j = 0; j < 4; ++j)
#pragma unroll
    for (int r = 0; r < 8; ++r) {
      int grow = rowBase + wvid * 16 + r + 8 * h16;
      if (grow >= M) continue;
      int gcol  = nBase + j * 16 + col0;
      float val = acc[j][r];
      if (MODE == 2) {
        out0[(size_t)grow * N + gcol] = val + bias[gcol];
      } else {
        float* dst = out0;
        int cc = gcol;
        if (MODE == 0 && gcol >= 768) { dst = out1; cc = gcol - 768; }
        int h  = cc >> 6, d = cc & 63;
        int bi = grow / N_TOK, n = grow % N_TOK;
        dst[(((size_t)bi * NH_ + h) * N_TOK + n) * HD_ + d] = val;
      }
    }
}

// ---------------------------------------------------------------------------
// Depthwise 3x3 conv (pad 1, stride s) over 32x32 body + LayerNorm over hd=64.
// One 64-thread block per output token; CLS token bypasses conv, gets LN.
// ---------------------------------------------------------------------------
__global__ __launch_bounds__(64) void pool_norm_kernel(
    const float* __restrict__ tin, const float* __restrict__ w,
    const float* __restrict__ g, const float* __restrict__ bb,
    float* __restrict__ tout, int stride, int outW, int nTok)
{
  int tok = blockIdx.x % nTok;
  int bh  = blockIdx.x / nTok;
  int c   = threadIdx.x;
  float y;
  if (tok == 0) {
    y = tin[((size_t)bh * N_TOK) * HD_ + c];
  } else {
    int oy = (tok - 1) / outW, ox = (tok - 1) % outW;
    float s = 0.f;
#pragma unroll
    for (int ky = 0; ky < 3; ++ky)
#pragma unroll
      for (int kx = 0; kx < 3; ++kx) {
        int iy = oy * stride + ky - 1;
        int ix = ox * stride + kx - 1;
        if (iy >= 0 && iy < 32 && ix >= 0 && ix < 32)
          s += w[c * 9 + ky * 3 + kx] *
               tin[((size_t)bh * N_TOK + 1 + iy * 32 + ix) * HD_ + c];
      }
    y = s;
  }
  // LayerNorm over 64 channels (2 waves -> shfl + LDS combine)
  __shared__ float part[2];
  int lane = c & 31, wid = c >> 5;
  float v = y;
#pragma unroll
  for (int m = 16; m >= 1; m >>= 1) v += __shfl_xor(v, m, 32);
  if (lane == 0) part[wid] = v;
  __syncthreads();
  float mean = (part[0] + part[1]) * (1.f / 64.f);
  __syncthreads();
  float d = y - mean;
  v = d * d;
#pragma unroll
  for (int m = 16; m >= 1; m >>= 1) v += __shfl_xor(v, m, 32);
  if (lane == 0) part[wid] = v;
  __syncthreads();
  float var = (part[0] + part[1]) * (1.f / 64.f);
  tout[((size_t)bh * nTok + tok) * HD_ + c] = d * rsqrtf(var + 1e-5f) * g[c] + bb[c];
}

// ---------------------------------------------------------------------------
// rel_h[b,h,q,ky] = q_row . rel_pos_h[qy - 2*ky + 30]; same for rel_w with qx.
// 32 threads per body token: lanes 0-15 -> rel_h, 16-31 -> rel_w.
// ---------------------------------------------------------------------------
__global__ __launch_bounds__(32) void rel_bias_kernel(
    const float* __restrict__ qp, const float* __restrict__ rph,
    const float* __restrict__ rpw, float* __restrict__ relh,
    float* __restrict__ relw)
{
  int body = blockIdx.x & 1023;
  int bh   = blockIdx.x >> 10;
  int qy = body >> 5, qx = body & 31;
  const float* qrow = qp + ((size_t)bh * N_TOK + 1 + body) * HD_;
  int l = threadIdx.x;
  int k = l & 15;
  int qpos = (l >= 16) ? qx : qy;
  int idx  = qpos - 2 * k + 30;   // _rel_idx(32,16)
  const float* R = ((l >= 16) ? rpw : rph) + idx * HD_;
  float s = 0.f;
#pragma unroll
  for (int c = 0; c < HD_; ++c) s += qrow[c] * R[c];
  float* dst = (l >= 16) ? relw : relh;
  dst[((size_t)bh * 1024 + body) * 16 + k] = s;
}

// ---------------------------------------------------------------------------
// Flash attention: 1 wave per (b*h, 16-row q tile). K/V in chunks of 32 rows.
// S = Q*K^T via 2x2 WMMA, rel bias + mask, online softmax (shfl row-reduce),
// P re-fragmented through LDS, O += P*V via 4 WMMA. Residual +q, store in
// projection-input layout [B, N, 768].
// ---------------------------------------------------------------------------
__global__ __launch_bounds__(32) void attention_kernel(
    const float* __restrict__ qp, const float* __restrict__ kp,
    const float* __restrict__ vp, const float* __restrict__ relh,
    const float* __restrict__ relw, float* __restrict__ ao)
{
  __shared__ _Float16 Qs[16][68];
  __shared__ _Float16 Ks[32][68];
  __shared__ _Float16 VTs[64][36];  // V transposed: [c][k_local]
  __shared__ _Float16 Ps[16][36];

  int qtile = blockIdx.x;
  int bh    = blockIdx.y;
  int bi    = bh / NH_, hh = bh % NH_;
  int tid   = threadIdx.x;
  const float scale = 0.125f;  // hd^-0.5

  // stage Q tile (16x64)
#pragma unroll
  for (int t = 0; t < 32; ++t) {
    int idx = tid + t * 32;
    int r = idx >> 6, c = idx & 63;
    int qn = qtile * 16 + r;
    float v = (qn < N_TOK) ? qp[((size_t)bh * N_TOK + qn) * HD_ + c] : 0.f;
    Qs[r][c] = (_Float16)v;
  }
  __syncthreads();
  v16h aq0 = load_a_frag(&Qs[0][0], 68, 0);
  v16h aq1 = load_a_frag(&Qs[0][0], 68, 32);

  v8f o[4];
  float rowmax[8], rowsum[8];
#pragma unroll
  for (int j = 0; j < 4; ++j)
#pragma unroll
    for (int e = 0; e < 8; ++e) o[j][e] = 0.f;
#pragma unroll
  for (int r = 0; r < 8; ++r) { rowmax[r] = -1e30f; rowsum[r] = 0.f; }

  int l    = tid & 31;
  int h16  = (l >> 4) & 1;
  int col0 = l & 15;

  for (int ck = 0; ck < 9; ++ck) {   // 9*32 = 288 >= 257
    int kc = ck * 32;
    // stage K (row-major) and V (transposed)
#pragma unroll
    for (int t = 0; t < 64; ++t) {
      int idx = tid + t * 32;
      int r = idx >> 6, c = idx & 63;
      int kn = kc + r;
      float kvv = (kn < KV_N) ? kp[((size_t)bh * KV_N + kn) * HD_ + c] : 0.f;
      float vvv = (kn < KV_N) ? vp[((size_t)bh * KV_N + kn) * HD_ + c] : 0.f;
      Ks[r][c]  = (_Float16)kvv;
      VTs[c][r] = (_Float16)vvv;
    }
    __syncthreads();

    float sv[2][8];
#pragma unroll
    for (int j = 0; j < 2; ++j) {
      v8f s;
#pragma unroll
      for (int e = 0; e < 8; ++e) s[e] = 0.f;
      v16h bk0 = load_b_frag(&Ks[j * 16][0], 68, 0);
      s = __builtin_amdgcn_wmma_f32_16x16x32_f16(false, aq0, false, bk0, (short)0, s, false, false);
      v16h bk1 = load_b_frag(&Ks[j * 16][0], 68, 32);
      s = __builtin_amdgcn_wmma_f32_16x16x32_f16(false, aq1, false, bk1, (short)0, s, false, false);
      int kn = kc + j * 16 + col0;
#pragma unroll
      for (int r = 0; r < 8; ++r) {
        int qn = qtile * 16 + r + 8 * h16;
        float val;
        if (kn >= KV_N) {
          val = -1e30f;  // masked column
        } else {
          val = s[r] * scale;
          if (kn >= 1 && qn >= 1 && qn < N_TOK) {  // body-body rel bias
            int km = kn - 1;
            val += relh[((size_t)bh * 1024 + (qn - 1)) * 16 + (km >> 4)] +
                   relw[((size_t)bh * 1024 + (qn - 1)) * 16 + (km & 15)];
          }
        }
        sv[j][r] = val;
      }
    }

    // online softmax update (row = r + 8*h16, spread over a 16-lane half)
#pragma unroll
    for (int r = 0; r < 8; ++r) {
      float m2 = fmaxf(sv[0][r], sv[1][r]);
#pragma unroll
      for (int msk = 8; msk >= 1; msk >>= 1) m2 = fmaxf(m2, __shfl_xor(m2, msk, 32));
      float nm    = fmaxf(rowmax[r], m2);
      float alpha = __expf(rowmax[r] - nm);
      rowmax[r]   = nm;
      float p0 = __expf(sv[0][r] - nm);
      float p1 = __expf(sv[1][r] - nm);
      sv[0][r] = p0; sv[1][r] = p1;
      float ps = p0 + p1;
#pragma unroll
      for (int msk = 8; msk >= 1; msk >>= 1) ps += __shfl_xor(ps, msk, 32);
      rowsum[r] = rowsum[r] * alpha + ps;
#pragma unroll
      for (int j = 0; j < 4; ++j) o[j][r] *= alpha;
    }

    // P (16x32) -> LDS, re-fragment as A matrix
#pragma unroll
    for (int r = 0; r < 8; ++r) {
      Ps[r + 8 * h16][col0]      = (_Float16)sv[0][r];
      Ps[r + 8 * h16][16 + col0] = (_Float16)sv[1][r];
    }
    __syncthreads();
    v16h pa = load_a_frag(&Ps[0][0], 36, 0);
#pragma unroll
    for (int j = 0; j < 4; ++j) {
      v16h bvf = load_b_frag(&VTs[j * 16][0], 36, 0);
      o[j] = __builtin_amdgcn_wmma_f32_16x16x32_f16(false, pa, false, bvf, (short)0, o[j], false, false);
    }
    __syncthreads();
  }

  // finalize: /rowsum, +residual q (body rows only), store proj-input layout
#pragma unroll
  for (int j = 0; j < 4; ++j)
#pragma unroll
    for (int r = 0; r < 8; ++r) {
      int qn = qtile * 16 + r + 8 * h16;
      if (qn >= N_TOK) continue;
      int c = j * 16 + col0;
      float val = o[j][r] / rowsum[r];
      if (qn >= 1) val += qp[((size_t)bh * N_TOK + qn) * HD_ + c];
      ao[((size_t)bi * N_TOK + qn) * DIM_ + hh * HD_ + c] = val;
    }
}

// ---------------------------------------------------------------------------
extern "C" void kernel_launch(void* const* d_in, const int* in_sizes, int n_in,
                              void* d_out, int out_size, void* d_ws, size_t ws_size,
                              hipStream_t stream)
{
  (void)in_sizes; (void)n_in; (void)out_size; (void)ws_size;
  const float* query     = (const float*)d_in[0];
  const float* x         = (const float*)d_in[1];
  const float* W_qkv     = (const float*)d_in[2];
  const float* W_qkv2    = (const float*)d_in[3];
  const float* wq        = (const float*)d_in[4];
  const float* gq        = (const float*)d_in[5];
  const float* bq        = (const float*)d_in[6];
  const float* wk        = (const float*)d_in[7];
  const float* gk        = (const float*)d_in[8];
  const float* bk        = (const float*)d_in[9];
  const float* wv        = (const float*)d_in[10];
  const float* gv        = (const float*)d_in[11];
  const float* bv        = (const float*)d_in[12];
  const float* W_proj    = (const float*)d_in[13];
  const float* b_proj    = (const float*)d_in[14];
  const float* rel_pos_h = (const float*)d_in[15];
  const float* rel_pos_w = (const float*)d_in[16];

  float* ws = (float*)d_ws;
  size_t off = 0;
  const size_t SZ_FULL = (size_t)B_ * NH_ * N_TOK * HD_;  // 12,595,200
  const size_t SZ_KV   = (size_t)B_ * NH_ * KV_N * HD_;   //  3,158,016
  const size_t SZ_REL  = (size_t)B_ * NH_ * 1024 * 16;    //  3,145,728
  const size_t SZ_AO   = (size_t)B_ * N_TOK * DIM_;       // 12,595,200
  float* q0 = ws + off; off += SZ_FULL;
  float* k0 = ws + off; off += SZ_FULL;
  float* v0 = ws + off; off += SZ_FULL;
  float* qp = ws + off; off += SZ_FULL;
  float* kp = ws + off; off += SZ_KV;
  float* vp = ws + off; off += SZ_KV;
  float* rh = ws + off; off += SZ_REL;
  float* rw = ws + off; off += SZ_REL;
  float* ao = ws + off; off += SZ_AO;

  // 1) k,v = x @ W_qkv[:, 768:2304]  -> per-head buffers
  wmma_gemm<0><<<dim3(129, 24), dim3(256), 0, stream>>>(
      x, W_qkv, nullptr, k0, v0, M_ROWS, 1536, 768, 2304, 768);
  // 2) q = query @ W_qkv2 -> per-head buffer
  wmma_gemm<1><<<dim3(129, 12), dim3(256), 0, stream>>>(
      query, W_qkv2, nullptr, q0, nullptr, M_ROWS, 768, 768, 768, 0);
  // 3) pooling + layernorm
  pool_norm_kernel<<<B_ * NH_ * N_TOK, dim3(64), 0, stream>>>(q0, wq, gq, bq, qp, 1, 32, N_TOK);
  pool_norm_kernel<<<B_ * NH_ * KV_N, dim3(64), 0, stream>>>(k0, wk, gk, bk, kp, 2, 16, KV_N);
  pool_norm_kernel<<<B_ * NH_ * KV_N, dim3(64), 0, stream>>>(v0, wv, gv, bv, vp, 2, 16, KV_N);
  // 4) decomposed rel-pos bias tables
  rel_bias_kernel<<<B_ * NH_ * 1024, dim3(32), 0, stream>>>(qp, rel_pos_h, rel_pos_w, rh, rw);
  // 5) attention + residual
  attention_kernel<<<dim3(65, B_ * NH_), dim3(32), 0, stream>>>(qp, kp, vp, rh, rw, ao);
  // 6) projection + bias -> d_out
  wmma_gemm<2><<<dim3(129, 12), dim3(256), 0, stream>>>(
      ao, W_proj, b_proj, (float*)d_out, nullptr, M_ROWS, 768, 768, 768, 0);
}
